// XuHawkesTorch_8847632629795
// MI455X (gfx1250) — matrix operand order repaired and checked by reading, not compiled, not run on hardware
//
#include <hip/hip_runtime.h>
#include <hip/hip_bf16.h>
#include <math.h>

// ---------------------------------------------------------------------------
// Hawkes log-likelihood on gfx1250 (MI455X).
// Exponential-decay factorization breaks the M=16384 sequential scan into:
//   (1) 64-step chunk-state recurrence (only serial part),
//   (2) bf16 WMMA skinny GEMM for cross-chunk intensities
//       (B panel staged in LDS via the Tensor Data Mover),
//   (3) O(M*CHUNK/2) within-chunk pair sums,
//   (4) column-sum based compensator (1^T * alpha * v == colsum . v).
// ---------------------------------------------------------------------------

typedef __attribute__((ext_vector_type(16))) __bf16       v16bf;
typedef __attribute__((ext_vector_type(8)))  __bf16       v8bf;
typedef __attribute__((ext_vector_type(8)))  float        v8f;
typedef __attribute__((ext_vector_type(4)))  unsigned int v4u;
typedef __attribute__((ext_vector_type(4)))  int          v4i;
typedef __attribute__((ext_vector_type(8)))  int          v8i;

#define D_DIM    2048
#define M_EV     16384
#define CHUNK    256
#define NCHUNK   64          // M_EV / CHUNK
#define CH_PER_G 16          // chunks per WMMA supergroup (N dimension)
#define NTILE    1024        // M_EV / 16
#define EPS_MU   1e-6f
#define EPS_LOG  1e-8f

__device__ __forceinline__ float softplusf(float x) {
    return fmaxf(x, 0.0f) + log1pf(__expf(-fabsf(x)));
}

// ---------------------------------------------------------------------------
// K0a: alpha = softplus(log_alpha) -> bf16 copy for WMMA; fused column sums.
// ---------------------------------------------------------------------------
__global__ void prep_alpha_kernel(const float* __restrict__ log_alpha,
                                  __bf16* __restrict__ alpha_bf16,
                                  float* __restrict__ colsum) {
    const int tid = threadIdx.x;
    const int r0  = blockIdx.x * 128;
    float csum[8];
#pragma unroll
    for (int q = 0; q < 8; ++q) csum[q] = 0.0f;
    for (int r = r0; r < r0 + 128; ++r) {
        const float* row  = log_alpha  + (size_t)r * D_DIM;
        __bf16*      orow = alpha_bf16 + (size_t)r * D_DIM;
#pragma unroll
        for (int q = 0; q < 8; ++q) {
            const int c = tid + q * 256;
            const float a = softplusf(row[c]);
            orow[c] = (__bf16)a;
            csum[q] += a;
        }
    }
#pragma unroll
    for (int q = 0; q < 8; ++q) atomicAdd(&colsum[tid + q * 256], csum[q]);
}

// ---------------------------------------------------------------------------
// K0b: mu = softplus(log_mu) + eps
// ---------------------------------------------------------------------------
__global__ void prep_mu_kernel(const float* __restrict__ log_mu,
                               float* __restrict__ mu) {
    const int j = blockIdx.x * blockDim.x + threadIdx.x;
    if (j < D_DIM) mu[j] = softplusf(log_mu[j]) + EPS_MU;
}

// ---------------------------------------------------------------------------
// K0c: contrib[j] = sum over events with mark j of (1 - exp(-(T - t_n)))
// ---------------------------------------------------------------------------
__global__ void prep_contrib_kernel(const float* __restrict__ t,
                                    const int* __restrict__ marks,
                                    const int* __restrict__ T_p,
                                    float* __restrict__ contrib) {
    const int n = blockIdx.x * blockDim.x + threadIdx.x;
    if (n < M_EV) {
        const float T = (float)T_p[0];
        atomicAdd(&contrib[marks[n]], 1.0f - __expf(t[n] - T));
    }
}

// ---------------------------------------------------------------------------
// K1: chunk-state recurrence. S^{(c)}[j] = sum_{m < c*CHUNK} e^{-(t_ref(c)-t_m)}
// Single block, 256 threads; thread owns marks [tid*8, tid*8+8).
// ---------------------------------------------------------------------------
__global__ void chunk_states_kernel(const float* __restrict__ t,
                                    const int* __restrict__ marks,
                                    __bf16* __restrict__ Sb) {
    __shared__ float t_l[CHUNK];
    __shared__ int   m_l[CHUNK];
    const int tid = threadIdx.x;
    float s[8];
#pragma unroll
    for (int r = 0; r < 8; ++r) s[r] = 0.0f;

    for (int c = 0; c < NCHUNK; ++c) {
#pragma unroll
        for (int r = 0; r < 8; ++r)
            Sb[(size_t)c * D_DIM + tid * 8 + r] = (__bf16)s[r];
        if (c == NCHUNK - 1) break;

        t_l[tid] = t[c * CHUNK + tid];
        m_l[tid] = marks[c * CHUNK + tid];
        __syncthreads();

        const float t_cur = t_l[0];
        const float t_nxt = t[(c + 1) * CHUNK];
        const float dec   = __expf(t_cur - t_nxt);
#pragma unroll
        for (int r = 0; r < 8; ++r) s[r] *= dec;
        for (int e = 0; e < CHUNK; ++e) {
            const int mk = m_l[e];
            if ((mk >> 3) == tid) s[mk & 7] += __expf(t_l[e] - t_nxt);
        }
        __syncthreads();
    }
}

// ---------------------------------------------------------------------------
// K2: WMMA cross-chunk GEMM.  One wave per 16-event tile, 4 waves/block.
// The block's shared B panel (16 chunk states x 2048 bf16 = 64 KB) is staged
// into LDS once with the Tensor Data Mover (TENSOR_LOAD_TO_LDS, 2-D D#),
// then all 4 waves read it via ds_load while A rows stream from L2.
// ---------------------------------------------------------------------------
__global__ void cross_gemm_kernel(const __bf16* __restrict__ alpha_bf16,
                                  const __bf16* __restrict__ Sb,
                                  const int* __restrict__ marks,
                                  float* __restrict__ G) {
    __shared__ __align__(16) __bf16 Bsh[CH_PER_G * D_DIM];   // 64 KB

    const int lane = threadIdx.x & 31;
    const int warp = threadIdx.x >> 5;
    const int tile = blockIdx.x * (blockDim.x >> 5) + warp;   // 0..NTILE-1
    const int grpb = (blockIdx.x * (blockDim.x >> 5)) >> 8;   // block's supergroup

    if (warp == 0) {
        // --- Tensor DMA: copy supergroup state panel Sb[grpb*16 .. +16) to LDS
        const unsigned long long ga =
            (unsigned long long)(uintptr_t)(Sb + (size_t)grpb * CH_PER_G * D_DIM);
        // LDS aperture maps addr[31:0] directly to the LDS byte address.
        const unsigned int lds_off = (unsigned int)(uintptr_t)(&Bsh[0]);
        // D# group0: count=1, lds_addr, global_addr[56:0], type=2 ("image")
        v4u g0 = { 1u,
                   lds_off,
                   (unsigned int)(ga & 0xFFFFFFFFu),
                   (unsigned int)(((ga >> 32) & 0x01FFFFFFu) | (2u << 30)) };
        // D# group1: data_size=2B; tensor 2048x16, tile 2048x16, stride0=2048
        v8i g1 = { (int)0x00010000,          // [17:16]=1 -> 2-byte elements
                   (int)(D_DIM << 16),       // tensor_dim0[15:0] in bits[63:48]
                   (int)(CH_PER_G << 16),    // tensor_dim1[15:0] in bits[95:80]
                   (int)(D_DIM << 16),       // tile_dim0 in bits[127:112]
                   (int)CH_PER_G,            // tile_dim1 in bits[143:128]
                   (int)D_DIM,               // tensor_dim0_stride low bits
                   0, 0 };
        v4i gz = { 0, 0, 0, 0 };
#if __has_include(<hip/amd_detail/amd_gfx1250_TDM.h>)
        v8i gz8 = { 0, 0, 0, 0, 0, 0, 0, 0 };
        __builtin_amdgcn_tensor_load_to_lds(g0, g1, gz, gz, gz8, 0);
#else
        __builtin_amdgcn_tensor_load_to_lds(g0, g1, gz, gz, 0);
#endif
        __builtin_amdgcn_s_wait_tensorcnt(0);
    }
    __syncthreads();

    const int m  = lane & 15;            // A row id / B column id
    const int hi = lane >> 4;            // K-half select
    const int ev = tile * 16 + m;
    const int row = marks[ev];

    const __bf16* Abase = alpha_bf16 + (size_t)row * D_DIM;
    const __bf16* Bl    = Bsh + (size_t)m * D_DIM;   // state (column) m of group
    const int ka = hi * 8;    // A: lanes 16-31 hold K {k+8..15, k+24..31}
    const int kb = hi * 8;    // B: lanes 16-31 hold K {k+8..15, k+24..31}

    v8f acc = {};
    for (int k = 0; k < D_DIM; k += 32) {
        v8bf a0 = *(const v8bf*)(Abase + k + ka);
        v8bf a1 = *(const v8bf*)(Abase + k + 16 + ka);
        v8bf b0 = *(const v8bf*)(Bl + k + kb);
        v8bf b1 = *(const v8bf*)(Bl + k + 16 + kb);
        v16bf A = __builtin_shufflevector(a0, a1, 0,1,2,3,4,5,6,7,8,9,10,11,12,13,14,15);
        v16bf B = __builtin_shufflevector(b0, b1, 0,1,2,3,4,5,6,7,8,9,10,11,12,13,14,15);
        acc = __builtin_amdgcn_wmma_f32_16x16x32_bf16(
            /*neg_a=*/false, A, /*neg_b=*/false, B,
            /*c_mod=*/(short)0, acc, /*reuse_a=*/false, /*reuse_b=*/false);
    }

    // C/D layout: lane<16 -> rows 0..7, lane>=16 -> rows 8..15; col = lane&15
    float* out = G + tile * 256;
#pragma unroll
    for (int r = 0; r < 8; ++r)
        out[(r + hi * 8) * 16 + m] = acc[r];
}

// ---------------------------------------------------------------------------
// K3: within-chunk pairwise term + combine + log-sum.  One block per chunk.
// Gathers the precomputed bf16 alpha (no softplus recompute).
// ---------------------------------------------------------------------------
__global__ void within_chunk_kernel(const float* __restrict__ t,
                                    const int* __restrict__ marks,
                                    const __bf16* __restrict__ alpha_bf16,
                                    const float* __restrict__ mu,
                                    const float* __restrict__ G,
                                    float* __restrict__ loglik) {
    __shared__ float t_l[CHUNK];
    __shared__ int   m_l[CHUNK];
    __shared__ float red[CHUNK];
    const int c   = blockIdx.x;
    const int tid = threadIdx.x;
    const int n   = c * CHUNK + tid;

    t_l[tid] = t[n];
    m_l[tid] = marks[n];
    __syncthreads();

    const int   d_n = m_l[tid];
    const float t_n = t_l[tid];
    const __bf16* arow = alpha_bf16 + (size_t)d_n * D_DIM;
    float W = 0.0f;
    for (int mm = 0; mm < tid; ++mm)
        W += (float)arow[m_l[mm]] * __expf(t_l[mm] - t_n);

    const int tile = n >> 4;
    const float Gv = G[tile * 256 + (n & 15) * 16 + ((n >> 8) & 15)];
    const float lam = mu[d_n] + __expf(t_l[0] - t_n) * Gv + W;
    red[tid] = logf(lam + EPS_LOG);
    __syncthreads();
    for (int sft = 128; sft > 0; sft >>= 1) {
        if (tid < sft) red[tid] += red[tid + sft];
        __syncthreads();
    }
    if (tid == 0) atomicAdd(loglik, red[0]);
}

// ---------------------------------------------------------------------------
// K4: out = loglik - [ sum(mu)*T + sum_j colsum[j]*contrib[j] ]   (beta = 1)
// ---------------------------------------------------------------------------
__global__ void finalize_kernel(const float* __restrict__ mu,
                                const float* __restrict__ colsum,
                                const float* __restrict__ contrib,
                                const float* __restrict__ loglik,
                                const int* __restrict__ T_p,
                                float* __restrict__ out) {
    __shared__ float red[256];
    const int tid = threadIdx.x;
    const float T = (float)T_p[0];
    float s = 0.0f;
    for (int j = tid; j < D_DIM; j += 256)
        s += mu[j] * T + colsum[j] * contrib[j];
    red[tid] = s;
    __syncthreads();
    for (int sft = 128; sft > 0; sft >>= 1) {
        if (tid < sft) red[tid] += red[tid + sft];
        __syncthreads();
    }
    if (tid == 0) out[0] = loglik[0] - red[0];
}

// ---------------------------------------------------------------------------
extern "C" void kernel_launch(void* const* d_in, const int* in_sizes, int n_in,
                              void* d_out, int out_size, void* d_ws, size_t ws_size,
                              hipStream_t stream) {
    const float* t_events  = (const float*)d_in[0];
    const int*   marks     = (const int*)  d_in[1];
    const int*   T_p       = (const int*)  d_in[2];   // python int scalar -> int32
    const float* log_mu    = (const float*)d_in[3];
    const float* log_alpha = (const float*)d_in[4];
    float* out = (float*)d_out;

    // workspace layout (~9.7 MB)
    char* ws = (char*)d_ws;
    size_t off = 0;
    __bf16* alpha_bf16 = (__bf16*)(ws + off); off += (size_t)D_DIM * D_DIM * 2;      // 8 MB
    float*  G          = (float*) (ws + off); off += (size_t)NTILE * 256 * 4;        // 1 MB
    __bf16* Sb         = (__bf16*)(ws + off); off += (size_t)NCHUNK * D_DIM * 2;     // 256 KB
    float*  colsum     = (float*) (ws + off); off += (size_t)D_DIM * 4;
    float*  mu         = (float*) (ws + off); off += (size_t)D_DIM * 4;
    float*  contrib    = (float*) (ws + off); off += (size_t)D_DIM * 4;
    float*  loglik     = (float*) (ws + off); off += 256;
    (void)ws_size; (void)in_sizes; (void)n_in; (void)out_size;

    // zero accumulators (colsum, mu(harmless), contrib, loglik) — capture-safe
    hipMemsetAsync(colsum, 0, (size_t)D_DIM * 4 * 3 + 256, stream);

    prep_alpha_kernel  <<<16,  256, 0, stream>>>(log_alpha, alpha_bf16, colsum);
    prep_mu_kernel     <<<8,   256, 0, stream>>>(log_mu, mu);
    prep_contrib_kernel<<<64,  256, 0, stream>>>(t_events, marks, T_p, contrib);
    chunk_states_kernel<<<1,   256, 0, stream>>>(t_events, marks, Sb);
    cross_gemm_kernel  <<<NTILE / 4, 128, 0, stream>>>(alpha_bf16, Sb, marks, G);
    within_chunk_kernel<<<NCHUNK, 256, 0, stream>>>(t_events, marks, alpha_bf16, mu, G, loglik);
    finalize_kernel    <<<1,   256, 0, stream>>>(mu, colsum, contrib, loglik, T_p, out);
}
